// IntegratedAutoEncoder_34626026341035
// MI455X (gfx1250) — matrix operand order repaired and checked by reading, not compiled, not run on hardware
//
#include <hip/hip_runtime.h>
#include <hip/hip_bf16.h>

// ---------------- problem constants (from reference) ----------------
constexpr int BS    = 4;
constexpr int Hh    = 80;
constexpr int Ww    = 80;
constexpr int C     = 256;
constexpr int NHEAD = 8;
constexpr int NPTS  = 4;
constexpr int FF    = 1024;
constexpr int HD    = C / NHEAD;   // 32 == wave32
constexpr int L     = Hh * Ww;     // 6400
constexpr int Mr    = BS * L;      // 25600 rows (multiple of 64)

typedef __attribute__((ext_vector_type(16))) _Float16 v16h;
typedef __attribute__((ext_vector_type(8)))  _Float16 v8h;
typedef __attribute__((ext_vector_type(4)))  _Float16 v4h;
typedef __attribute__((ext_vector_type(8)))  float    v8f;
typedef __attribute__((ext_vector_type(4)))  float    v4f;

// ---------------- fused 1x1 conv + BN -> src [Mr, C] ----------------
__global__ void __launch_bounds__(256)
k_src(const float* __restrict__ feats, const float* __restrict__ conv_w,
      const float* __restrict__ bn_g, const float* __restrict__ bn_b,
      const float* __restrict__ bn_m, const float* __restrict__ bn_v,
      float* __restrict__ src)
{
    int idx = blockIdx.x * 256 + threadIdx.x;     // over Mr*C
    int c = idx & (C - 1);
    int m = idx >> 8;
    float s  = bn_g[c] * rsqrtf(bn_v[c] + 1e-5f);
    float sa = conv_w[c] * s;
    float sb = bn_b[c] - bn_m[c] * s;
    src[idx] = feats[m] * sa + sb;
}

// ------- weight prep: [K,N] f32 -> transposed [N,K] f16 --------------
__global__ void __launch_bounds__(256)
k_wcvt(const float* __restrict__ in, _Float16* __restrict__ out, int K, int N)
{
    int id = blockIdx.x * 256 + threadIdx.x;      // over N*K/4
    int kq = id % (K >> 2);
    int n  = id / (K >> 2);
    if (n >= N) return;
    int k = kq << 2;
    v4h h;
    #pragma unroll
    for (int j = 0; j < 4; ++j) h[j] = (_Float16)in[(size_t)(k + j) * N + n];
    *(v4h*)(out + (size_t)n * K + k) = h;
}

// ---------------- WMMA f16 GEMM: out = A[MxK] * B[KxN] + bias --------
// B supplied pre-transposed+converted: BT[N][K] f16 (L2 resident).
// Block tile 64x64, 8 waves; wave (wm, wn) owns rows wm*16.., cols wn*32..
// A tile staged in LDS in WMMA-fragment-ready layout so the per-wave
// fragment load is two ds_load_b128; B fragments are per-lane K-contiguous
// rows of BT -> two global_load_b128 each, no LDS staging for B.
// Columns past N are handled by CLAMPING the load column (in-bounds reads,
// garbage results) and guarding only the epilogue stores -> branch-free
// inner loop, no exec-mask juggling, no zero-fill.
template <typename TA>
__global__ void __launch_bounds__(256)
gemm_wmma(const TA* __restrict__ A, const _Float16* __restrict__ BT,
          const float* __restrict__ bias, float* __restrict__ outF,
          _Float16* __restrict__ outH, int M, int N, int K, int relu)
{
    // fragment-ready A: [row-group][lane][16 halves] (+8 pad for alignment)
    __shared__ _Float16 AsF[4][32][24];

    const int t    = threadIdx.x;
    const int m0   = blockIdx.x * 64;
    const int n0   = blockIdx.y * 64;
    const int wave = t >> 5;
    const int lane = t & 31;
    const int wm   = wave & 3;        // 16-row group
    const int wn   = wave >> 2;       // 32-col group
    const int hl   = lane & 15;
    const bool lo  = lane < 16;
    const int kb   = lo ? 0 : 8;      // B fragment K base for this lane

    const int colA = n0 + wn * 32 + hl;
    const int colB = colA + 16;
    // clamped columns: OOB columns read valid (last) row of BT; their
    // results are garbage but never stored.
    const int colAc = colA < N ? colA : (N - 1);
    const int colBc = colB < N ? colB : (N - 1);
    const _Float16* bpA = BT + (size_t)colAc * K + kb;
    const _Float16* bpB = BT + (size_t)colBc * K + kb;

    v8f acc0 = {};
    v8f acc1 = {};

    for (int k0 = 0; k0 < K; k0 += 32) {
        // ---- stage A tile (64 rows x 32 K) into fragment layout ----
        if constexpr (sizeof(TA) == 4) {
            #pragma unroll
            for (int i = 0; i < 2; ++i) {
                int idx = t + i * 256;          // 512 float4 chunks
                int r   = idx >> 3;
                int c   = (idx & 7) << 2;       // K offset, 4-aligned
                v4f f = *(const v4f*)((const float*)A + (size_t)(m0 + r) * K + k0 + c);
                v4h h;
                h[0] = (_Float16)f[0]; h[1] = (_Float16)f[1];
                h[2] = (_Float16)f[2]; h[3] = (_Float16)f[3];
                int blk = c >> 3, cb = c & 7;   // K block (8-wide), offset in block
                int dl  = (r & 15) + ((blk & 1) << 4);
                int de  = ((blk >> 1) << 3) + cb;
                *(v4h*)&AsF[r >> 4][dl][de] = h;
            }
        } else {
            int r = t >> 2;                      // 256 v8h chunks
            int c8 = t & 3;
            int c  = c8 << 3;                    // K offset, 8-aligned
            v8h h = *(const v8h*)((const _Float16*)A + (size_t)(m0 + r) * K + k0 + c);
            int dl = (r & 15) + ((c8 & 1) << 4);
            int de = (c8 >> 1) << 3;
            *(v8h*)&AsF[r >> 4][dl][de] = h;
        }
        __syncthreads();

        // ---- A fragment: two 16B LDS loads ----
        v8h alo = *(const v8h*)&AsF[wm][lane][0];
        v8h ahi = *(const v8h*)&AsF[wm][lane][8];
        v16h a = __builtin_shufflevector(alo, ahi,
                   0,1,2,3,4,5,6,7,8,9,10,11,12,13,14,15);

        // ---- B fragments straight from transposed f16 weights ----
        v8h b0lo = *(const v8h*)(bpA + k0);
        v8h b0hi = *(const v8h*)(bpA + k0 + 16);
        v8h b1lo = *(const v8h*)(bpB + k0);
        v8h b1hi = *(const v8h*)(bpB + k0 + 16);
        v16h b0 = __builtin_shufflevector(b0lo, b0hi,
                    0,1,2,3,4,5,6,7,8,9,10,11,12,13,14,15);
        v16h b1 = __builtin_shufflevector(b1lo, b1hi,
                    0,1,2,3,4,5,6,7,8,9,10,11,12,13,14,15);

        acc0 = __builtin_amdgcn_wmma_f32_16x16x32_f16(false, a, false, b0,
                                                      (short)0, acc0, false, false);
        acc1 = __builtin_amdgcn_wmma_f32_16x16x32_f16(false, a, false, b1,
                                                      (short)0, acc1, false, false);
        __syncthreads();
    }

    // D layout: VGPR i -> row (lo ? i : 8+i), col = lane%16
    const int rbase = m0 + wm * 16 + (lo ? 0 : 8);
    #pragma unroll
    for (int i = 0; i < 8; ++i) {
        int row = rbase + i;
        if (colA < N) {
            float r0 = acc0[i] + bias[colA];
            if (relu) r0 = fmaxf(r0, 0.f);
            if (outF) outF[(size_t)row * N + colA] = r0;
            if (outH) outH[(size_t)row * N + colA] = (_Float16)r0;
        }
        if (colB < N) {
            float r1 = acc1[i] + bias[colB];
            if (relu) r1 = fmaxf(r1, 0.f);
            if (outF) outF[(size_t)row * N + colB] = r1;
            if (outH) outH[(size_t)row * N + colB] = (_Float16)r1;
        }
    }
}

// ---------------- softmax over NPTS=4 per (m, head) ------------------
__global__ void __launch_bounds__(256)
k_softmax(float* __restrict__ aw, int total)
{
    int id = blockIdx.x * 256 + threadIdx.x;   // total = Mr*NHEAD
    if (id >= total) return;
    float* p = aw + (size_t)id * 4;
    float a0 = p[0], a1 = p[1], a2 = p[2], a3 = p[3];
    float mx = fmaxf(fmaxf(a0, a1), fmaxf(a2, a3));
    float e0 = expf(a0 - mx), e1 = expf(a1 - mx), e2 = expf(a2 - mx), e3 = expf(a3 - mx);
    float inv = 1.f / (e0 + e1 + e2 + e3);
    p[0] = e0 * inv; p[1] = e1 * inv; p[2] = e2 * inv; p[3] = e3 * inv;
}

// ---------------- deformable sampling: wave per (m, head), lane = hd --
__device__ __forceinline__
float samp(const float* __restrict__ v, int b, int y, int x, int colofs)
{
    if ((unsigned)y >= (unsigned)Hh || (unsigned)x >= (unsigned)Ww) return 0.f;
    return v[((size_t)b * L + (size_t)y * Ww + x) * C + colofs];
}

__global__ void __launch_bounds__(256)
k_msdeform(const float* __restrict__ v, const float* __restrict__ off,
           const float* __restrict__ aw, float* __restrict__ attn)
{
    const int m    = blockIdx.x;          // b*L + l
    const int h    = threadIdx.x >> 5;    // head = wave id
    const int lane = threadIdx.x & 31;    // hd
    const int b  = m / L;
    const int l  = m - b * L;
    const int ix = l % Ww;
    const int iy = l / Ww;
    const float* offp = off + (size_t)m * (NHEAD * NPTS * 2) + h * (NPTS * 2);
    const float* awp  = aw  + (size_t)m * (NHEAD * NPTS)     + h * NPTS;
    const int colofs  = h * HD + lane;

    float res = 0.f;
    #pragma unroll
    for (int p = 0; p < NPTS; ++p) {
        // pixel-center ref grid => sample pixel = (ix + off_x, iy + off_y)
        float px = (float)ix + offp[p * 2 + 0];
        float py = (float)iy + offp[p * 2 + 1];
        float w  = awp[p];
        float x0f = floorf(px), y0f = floorf(py);
        float lx = px - x0f, ly = py - y0f;
        int x0 = (int)x0f, y0 = (int)y0f;
        float s = samp(v, b, y0,     x0,     colofs) * (1.f - lx) * (1.f - ly)
                + samp(v, b, y0,     x0 + 1, colofs) * lx         * (1.f - ly)
                + samp(v, b, y0 + 1, x0,     colofs) * (1.f - lx) * ly
                + samp(v, b, y0 + 1, x0 + 1, colofs) * lx         * ly;
        res += w * s;
    }
    attn[(size_t)m * C + colofs] = res;
}

// ---------------- out = residual + LayerNorm(x); wave per row --------
__global__ void __launch_bounds__(256)
k_add_ln(const float* __restrict__ residual, const float* __restrict__ x,
         const float* __restrict__ g, const float* __restrict__ b,
         float* __restrict__ out)
{
    const int row  = blockIdx.x * 8 + (threadIdx.x >> 5);
    const int lane = threadIdx.x & 31;
    const float* xr = x + (size_t)row * C;
    float vals[8];
    float s = 0.f, s2 = 0.f;
    #pragma unroll
    for (int j = 0; j < 8; ++j) {
        float t = xr[j * 32 + lane];
        vals[j] = t; s += t; s2 += t * t;
    }
    #pragma unroll
    for (int o = 16; o > 0; o >>= 1) {
        s  += __shfl_xor(s,  o, 32);
        s2 += __shfl_xor(s2, o, 32);
    }
    float mean = s * (1.f / C);
    float var  = s2 * (1.f / C) - mean * mean;
    float rstd = rsqrtf(var + 1e-5f);
    #pragma unroll
    for (int j = 0; j < 8; ++j) {
        int c = j * 32 + lane;
        out[(size_t)row * C + c] =
            residual[(size_t)row * C + c] + (vals[j] - mean) * rstd * g[c] + b[c];
    }
}

// ---------------- [Mr, C] -> [BS, C, L] LDS-tiled transpose ----------
__global__ void __launch_bounds__(256)
k_transpose(const float* __restrict__ in, float* __restrict__ out)
{
    __shared__ float tile[32][33];
    const int l0 = blockIdx.x * 32;
    const int c0 = blockIdx.y * 32;
    const int b  = blockIdx.z;
    #pragma unroll
    for (int j = 0; j < 4; ++j) {
        int l = l0 + threadIdx.y + j * 8;
        tile[threadIdx.y + j * 8][threadIdx.x] =
            in[((size_t)b * L + l) * C + c0 + threadIdx.x];
    }
    __syncthreads();
    #pragma unroll
    for (int j = 0; j < 4; ++j) {
        int c = c0 + threadIdx.y + j * 8;
        out[((size_t)b * C + c) * L + l0 + threadIdx.x] =
            tile[threadIdx.x][threadIdx.y + j * 8];
    }
}

// ---------------- launch ---------------------------------------------
extern "C" void kernel_launch(void* const* d_in, const int* in_sizes, int n_in,
                              void* d_out, int out_size, void* d_ws, size_t ws_size,
                              hipStream_t stream)
{
    const float* feats  = (const float*)d_in[0];
    const float* conv_w = (const float*)d_in[1];
    const float* bn_g   = (const float*)d_in[2];
    const float* bn_b   = (const float*)d_in[3];
    const float* bn_m   = (const float*)d_in[4];
    const float* bn_v   = (const float*)d_in[5];
    const float* w_val  = (const float*)d_in[6];
    const float* b_val  = (const float*)d_in[7];
    const float* w_off  = (const float*)d_in[8];
    const float* b_off  = (const float*)d_in[9];
    const float* w_aw   = (const float*)d_in[10];
    const float* b_aw   = (const float*)d_in[11];
    const float* w_out  = (const float*)d_in[12];
    const float* b_out  = (const float*)d_in[13];
    const float* ln1_g  = (const float*)d_in[14];
    const float* ln1_b  = (const float*)d_in[15];
    const float* w_ff1  = (const float*)d_in[16];
    const float* b_ff1  = (const float*)d_in[17];
    const float* w_ff2  = (const float*)d_in[18];
    const float* b_ff2  = (const float*)d_in[19];
    const float* ln2_g  = (const float*)d_in[20];
    const float* ln2_b  = (const float*)d_in[21];

    // workspace layout (floats), with liveness-based reuse
    float* ws    = (float*)d_ws;
    float* src   = ws;                              // Mr*C
    float* vbuf  = src  + (size_t)Mr * C;           // Mr*C
    float* offb  = vbuf + (size_t)Mr * C;           // Mr*64
    float* awb   = offb + (size_t)Mr * 64;          // Mr*32
    float* attn  = awb  + (size_t)Mr * 32;          // Mr*C
    _Float16* ffh = (_Float16*)(attn + (size_t)Mr * C);  // Mr*FF halves
    // transposed f16 weight pool
    _Float16* wvalT  = ffh    + (size_t)Mr * FF;    // C*C
    _Float16* woffT  = wvalT  + (size_t)C * C;      // C*64
    _Float16* wawT   = woffT  + (size_t)C * 64;     // C*32
    _Float16* woutT  = wawT   + (size_t)C * 32;     // C*C
    _Float16* wff1T  = woutT  + (size_t)C * C;      // C*FF
    _Float16* wff2T  = wff1T  + (size_t)C * FF;     // FF*C
    float* attnp = vbuf;   // v dead after msdeform
    float* src2  = attn;   // attn dead after out-proj GEMM
    float* ffo   = vbuf;   // attnp dead after first add_ln
    float* src3  = src;    // src (residual) dead after first add_ln

    // 0) weight convert+transpose (tiny, L2 resident afterwards)
    k_wcvt<<<(C * C / 4) / 256, 256, 0, stream>>>(w_val, wvalT, C, C);
    k_wcvt<<<(C * 64 / 4) / 256, 256, 0, stream>>>(w_off, woffT, C, 64);
    k_wcvt<<<(C * 32 / 4) / 256, 256, 0, stream>>>(w_aw,  wawT,  C, 32);
    k_wcvt<<<(C * C / 4) / 256, 256, 0, stream>>>(w_out, woutT, C, C);
    k_wcvt<<<(C * FF / 4) / 256, 256, 0, stream>>>(w_ff1, wff1T, C, FF);
    k_wcvt<<<(FF * C / 4) / 256, 256, 0, stream>>>(w_ff2, wff2T, FF, C);

    // 1) fused conv+BN
    k_src<<<(Mr * C) / 256, 256, 0, stream>>>(feats, conv_w, bn_g, bn_b, bn_m, bn_v, src);

    dim3 gC(Mr / 64, C / 64);       // N=256 GEMMs
    dim3 g1(Mr / 64, 1);            // N<=64 GEMMs
    // 2) value projection
    gemm_wmma<float><<<gC, 256, 0, stream>>>(src, wvalT, b_val, vbuf, nullptr, Mr, C, C, 0);
    // 3) sampling offsets
    gemm_wmma<float><<<g1, 256, 0, stream>>>(src, woffT, b_off, offb, nullptr, Mr, NHEAD * NPTS * 2, C, 0);
    // 4) attention-weight logits
    gemm_wmma<float><<<g1, 256, 0, stream>>>(src, wawT, b_aw, awb, nullptr, Mr, NHEAD * NPTS, C, 0);
    // 5) softmax over points
    k_softmax<<<(Mr * NHEAD) / 256, 256, 0, stream>>>(awb, Mr * NHEAD);
    // 6) deformable bilinear gather
    k_msdeform<<<Mr, 256, 0, stream>>>(vbuf, offb, awb, attn);
    // 7) output projection
    gemm_wmma<float><<<gC, 256, 0, stream>>>(attn, woutT, b_out, attnp, nullptr, Mr, C, C, 0);
    // 8) src2 = src + LN(attnp)
    k_add_ln<<<Mr / 8, 256, 0, stream>>>(src, attnp, ln1_g, ln1_b, src2);
    // 9) ff1 + ReLU (f16 hidden halves the dominant traffic term)
    gemm_wmma<float><<<dim3(Mr / 64, FF / 64), 256, 0, stream>>>(src2, wff1T, b_ff1, nullptr, ffh, Mr, FF, C, 1);
    // 10) ff2 (f16 A path)
    gemm_wmma<_Float16><<<gC, 256, 0, stream>>>(ffh, wff2T, b_ff2, ffo, nullptr, Mr, C, FF, 0);
    // 11) src3 = src2 + LN(ffo)
    k_add_ln<<<Mr / 8, 256, 0, stream>>>(src2, ffo, ln2_g, ln2_b, src3);
    // 12) [Mr,C] -> [BS,C,H,W]
    dim3 tg(L / 32, C / 32, BS), tb(32, 8);
    k_transpose<<<tg, tb, 0, stream>>>(src3, (float*)d_out);
}
